// MultiHeadSelfAttention_65429531788008
// MI455X (gfx1250) — matrix operand order repaired and checked by reading, not compiled, not run on hardware
//
#include <hip/hip_runtime.h>
#include <hip/hip_bf16.h>
#include <stdint.h>

// ---- problem constants (match reference) ----
#define BATCH   2
#define SEQ     2048
#define DMODEL  1024
#define HEADS   16
#define HDIM    64
#define MROWS   (BATCH * SEQ)          // 4096
#define SOFTMAX_SCALE 0.125f           // 1/sqrt(64)
#define GK      DMODEL                 // GEMM K (compile-time: all GEMMs are K=1024)
#define NSTEPS  (GK / 32)

typedef __attribute__((ext_vector_type(16))) __bf16 v16bf;
typedef __attribute__((ext_vector_type(8)))  __bf16 v8bf;
typedef __attribute__((ext_vector_type(8)))  float  v8f;

// ---------------- CDNA5 async copy helpers ----------------
__device__ __forceinline__ void async_copy_b128(unsigned lds_off, const void* gptr) {
    asm volatile("global_load_async_to_lds_b128 %0, %1, off"
                 :: "v"(lds_off), "v"(gptr)
                 : "memory");
}
__device__ __forceinline__ void wait_async_0() {
    asm volatile("s_wait_asynccnt 0x0" ::: "memory");
}
__device__ __forceinline__ unsigned lds_lo32(const void* p) {
    // generic shared pointer: low 32 bits are the LDS byte address
    return (unsigned)(uintptr_t)p;
}

// ---------------- fragment loaders ----------------
__device__ __forceinline__ v16bf load_afrag(const __bf16* p) {   // p = row + k0 + koffA
    v8bf lo = *(const v8bf*)(p);
    v8bf hi = *(const v8bf*)(p + 16);
    v16bf a;
    #pragma unroll
    for (int j = 0; j < 8; ++j) { a[j] = lo[j]; a[8 + j] = hi[j]; }
    return a;
}
__device__ __forceinline__ v16bf load_bfrag_lds(const __bf16* p) {
    v8bf lo = *(const v8bf*)(p);
    v8bf hi = *(const v8bf*)(p + 16);
    v16bf b;
    #pragma unroll
    for (int j = 0; j < 8; ++j) { b[j] = lo[j]; b[8 + j] = hi[j]; }
    return b;
}

// ---------------- reductions ----------------
__device__ __forceinline__ float grp16_max(float v) {
    v = fmaxf(v, __shfl_xor(v, 1));
    v = fmaxf(v, __shfl_xor(v, 2));
    v = fmaxf(v, __shfl_xor(v, 4));
    v = fmaxf(v, __shfl_xor(v, 8));
    return v;
}
__device__ __forceinline__ float grp16_sum(float v) {
    v += __shfl_xor(v, 1);
    v += __shfl_xor(v, 2);
    v += __shfl_xor(v, 4);
    v += __shfl_xor(v, 8);
    return v;
}

// ---------------- conversion kernels ----------------
__global__ void cvt_f32_to_bf16(const float* __restrict__ src, __bf16* __restrict__ dst, int n) {
    int i = blockIdx.x * blockDim.x + threadIdx.x;
    if (i < n) dst[i] = (__bf16)src[i];
}
// WT[n*K + k] = (bf16) W[k*N + n]
__global__ void cvt_transpose_bf16(const float* __restrict__ W, __bf16* __restrict__ WT, int K, int N) {
    int i = blockIdx.x * blockDim.x + threadIdx.x;
    if (i >= K * N) return;
    int n = i % N, k = i / N;
    WT[(size_t)n * K + k] = (__bf16)W[(size_t)k * N + n];
}
__global__ void zero_f32(float* __restrict__ p, int n) {
    int i = blockIdx.x * blockDim.x + threadIdx.x;
    if (i < n) p[i] = 0.0f;
}

// ---------------- bf16 WMMA GEMM with async-LDS B staging ----------------
// C[M,N] = A[M,1024] @ BT[N,1024]^T + bias[N]
// mode 0: f32 row-major Cf;  mode 1: bf16 per-head Cph[((b*H+h)*T + t)*64 + d]
// Block = 8 waves, each wave computes 32(M) x 64(N) -> block tile 256 x 64.
// B k-step tile (64 rows x 64B = 4KB) staged to LDS: one async b128 per thread,
// double buffered, ONE barrier per k-step (prefetch issued after the barrier).
// k-loop kept rolled (unroll 1) so accumulators stay pinned (no C->D moves, no
// WMMA->VALU hazard nops).
#define BROW 40               // LDS row stride in bf16 elems (80B = 20 banks)
__global__ __launch_bounds__(256)
void gemm_bf16_wmma(const __bf16* __restrict__ A, const __bf16* __restrict__ BT,
                    const float* __restrict__ bias,
                    float* __restrict__ Cf, __bf16* __restrict__ Cph,
                    int N, int mode) {
    __shared__ __bf16 Bls[2][64 * BROW];

    const int tid  = threadIdx.x;
    const int lane = tid & 31;
    const int wave = tid >> 5;
    const int half = lane >> 4;
    const int r    = lane & 15;
    const int m0   = blockIdx.y * 256 + wave * 32;
    const int n0   = blockIdx.x * 64;

    // async staging assignment: thread -> (B row n, 16B chunk)
    const int cn = tid >> 2;                 // 0..63
    const int cc = tid & 3;                  // 0..3
    const __bf16* gB = BT + (size_t)(n0 + cn) * GK + cc * 8;
    const unsigned ldsdst0 = lds_lo32(&Bls[0][0]) + (unsigned)(cn * (BROW * 2) + cc * 16);
    const unsigned ldsdst1 = ldsdst0 + (unsigned)(64 * BROW * 2);

    v8f acc[2][4] = {};
    const __bf16* Arow0 = A + (size_t)(m0 + r) * GK;
    const __bf16* Arow1 = A + (size_t)(m0 + 16 + r) * GK;
    const int koffA = half ? 8 : 0;          // A frag: lanes<16 K{0..7,16..23}, else {8..15,24..31}
    const int koffB = half ? 8 : 0;          // B frag: lanes<16 K{0..7,16..23}, else {8..15,24..31}

    async_copy_b128(ldsdst0, gB);            // prologue: stage k-step 0

    #pragma unroll 1
    for (int step = 0; step < NSTEPS; ++step) {
        const int k0  = step * 32;
        const int cur = step & 1;

        wait_async_0();                      // own buf[step] writes complete
        __syncthreads();                     // all waves: buf[step] written, buf[step-1] reads done
        if (step + 1 < NSTEPS)               // prefetch buf[step+1] (reuses buf[step-1])
            async_copy_b128(cur ? ldsdst0 : ldsdst1, gB + k0 + 32);

        const __bf16* bbase = &Bls[cur][0] + koffB;   // uniform scalar select of buffer
        v16bf a0 = load_afrag(Arow0 + k0 + koffA);
        v16bf a1 = load_afrag(Arow1 + k0 + koffA);
        #pragma unroll
        for (int t = 0; t < 4; ++t) {
            v16bf b = load_bfrag_lds(bbase + (t * 16 + r) * BROW);
            acc[0][t] = __builtin_amdgcn_wmma_f32_16x16x32_bf16(false, a0, false, b, (short)0, acc[0][t], false, false);
            acc[1][t] = __builtin_amdgcn_wmma_f32_16x16x32_bf16(false, a1, false, b, (short)0, acc[1][t], false, false);
        }
    }

    #pragma unroll
    for (int g = 0; g < 2; ++g) {
        #pragma unroll
        for (int t = 0; t < 4; ++t) {
            const int col = n0 + t * 16 + r;
            const float bv = bias[col];
            #pragma unroll
            for (int i = 0; i < 8; ++i) {
                const int row = m0 + g * 16 + i + half * 8;
                const float val = acc[g][t][i] + bv;
                if (mode == 0) {
                    Cf[(size_t)row * N + col] = val;
                } else {
                    const int b = row >> 11, tt = row & (SEQ - 1);
                    const int h = col >> 6,  d  = col & (HDIM - 1);
                    Cph[(((size_t)(b * HEADS + h) * SEQ) + tt) * HDIM + d] = (__bf16)val;
                }
            }
        }
    }
}

// ---------------- attention column-sum kernel ----------------
// One wave per (b, h, 16-row tk block). S[tk,tq] = K_h . Q_h^T via WMMA tiles.
// Pass 1: per-lane online max/sum (shuffle-free hot loop), merged once at end.
// Pass 2: recompute tiles, accumulate exp((s-m)*scale)/Z into colsum[b,h,tq].
__device__ __forceinline__ v8f score_tile(const __bf16* __restrict__ Q,
                                          v16bf a0, v16bf a1,
                                          int tq0, int r, int half) {
    const __bf16* Bp = Q + (size_t)(tq0 + r) * HDIM + (half ? 16 : 0);
    v16bf b0 = *(const v16bf*)(Bp);        // d 0..31
    v16bf b1 = *(const v16bf*)(Bp + 32);   // d 32..63
    v8f c = {};
    c = __builtin_amdgcn_wmma_f32_16x16x32_bf16(false, a0, false, b0, (short)0, c, false, false);
    c = __builtin_amdgcn_wmma_f32_16x16x32_bf16(false, a1, false, b1, (short)0, c, false, false);
    return c;
}

__global__ __launch_bounds__(32)
void attn_colsum(const __bf16* __restrict__ Qh, const __bf16* __restrict__ Kh,
                 float* __restrict__ colsum) {
    const int lane = threadIdx.x & 31;
    const int half = lane >> 4;
    const int r    = lane & 15;
    const int tk0  = blockIdx.x * 16;
    const int bh   = blockIdx.z * HEADS + blockIdx.y;

    const __bf16* Q  = Qh + (size_t)bh * SEQ * HDIM;
    const __bf16* Kp = Kh + (size_t)bh * SEQ * HDIM;

    v16bf a0, a1;
    {
        const __bf16* Arow = Kp + (size_t)(tk0 + r) * HDIM;
        const int koff = half ? 16 : 0;
        v8bf x0 = *(const v8bf*)(Arow + koff);
        v8bf x1 = *(const v8bf*)(Arow + 8 + koff);
        v8bf x2 = *(const v8bf*)(Arow + 32 + koff);
        v8bf x3 = *(const v8bf*)(Arow + 40 + koff);
        #pragma unroll
        for (int j = 0; j < 8; ++j) {
            a0[j] = x0[j]; a0[8 + j] = x1[j];
            a1[j] = x2[j]; a1[8 + j] = x3[j];
        }
    }

    float m[8], s[8];
    #pragma unroll
    for (int i = 0; i < 8; ++i) { m[i] = -1.0e30f; s[i] = 0.0f; }

    #pragma unroll 1
    for (int tq0 = 0; tq0 < SEQ; tq0 += 16) {
        v8f c = score_tile(Q, a0, a1, tq0, r, half);
        #pragma unroll
        for (int i = 0; i < 8; ++i) {
            const float mn = fmaxf(m[i], c[i]);
            s[i] = s[i] * __expf((m[i] - mn) * SOFTMAX_SCALE)
                 + __expf((c[i] - mn) * SOFTMAX_SCALE);
            m[i] = mn;
        }
    }
    float rinv[8];
    #pragma unroll
    for (int i = 0; i < 8; ++i) {
        const float mg = grp16_max(m[i]);
        const float sg = grp16_sum(s[i] * __expf((m[i] - mg) * SOFTMAX_SCALE));
        m[i] = mg;
        rinv[i] = 1.0f / sg;
    }

    float* csrow = colsum + (size_t)bh * SEQ;
    #pragma unroll 1
    for (int tq0 = 0; tq0 < SEQ; tq0 += 16) {
        v8f c = score_tile(Q, a0, a1, tq0, r, half);
        float colpart = 0.0f;
        #pragma unroll
        for (int i = 0; i < 8; ++i)
            colpart += __expf((c[i] - m[i]) * SOFTMAX_SCALE) * rinv[i];
        colpart += __shfl_xor(colpart, 16);
        if (lane < 16) atomicAdd(csrow + tq0 + r, colpart);
    }
}

// ---------------- elementwise mix: mixed = V * colsum ----------------
__global__ void mix_kernel(const float* __restrict__ V, const float* __restrict__ colsum,
                           __bf16* __restrict__ mixed, int total) {
    int i = blockIdx.x * blockDim.x + threadIdx.x;
    if (i >= total) return;
    const int h = (i >> 6) & (HEADS - 1);
    const int t = (i >> 10) & (SEQ - 1);
    const int b = i >> 21;
    mixed[i] = (__bf16)(V[i] * colsum[((size_t)(b * HEADS + h)) * SEQ + t]);
}

// ---------------- launch ----------------
static inline size_t align_up(size_t v, size_t a) { return (v + a - 1) & ~(a - 1); }

extern "C" void kernel_launch(void* const* d_in, const int* in_sizes, int n_in,
                              void* d_out, int out_size, void* d_ws, size_t ws_size,
                              hipStream_t stream) {
    const float* x  = (const float*)d_in[0];
    const float* Wq = (const float*)d_in[1];
    const float* bq = (const float*)d_in[2];
    const float* Wk = (const float*)d_in[3];
    const float* bk = (const float*)d_in[4];
    const float* Wv = (const float*)d_in[5];
    const float* bv = (const float*)d_in[6];
    const float* Wo = (const float*)d_in[7];
    const float* bo = (const float*)d_in[8];

    char* ws = (char*)d_ws;
    size_t off = 0;
    auto take = [&](size_t bytes) { size_t o = off; off = align_up(off + bytes, 256); return (void*)(ws + o); };

    const size_t nX = (size_t)MROWS * DMODEL;
    const size_t nW = (size_t)DMODEL * DMODEL;
    __bf16* Xbf   = (__bf16*)take(nX * 2);
    __bf16* WqT   = (__bf16*)take(nW * 2);
    __bf16* WkT   = (__bf16*)take(nW * 2);
    __bf16* WvT   = (__bf16*)take(nW * 2);
    __bf16* WoT   = (__bf16*)take(nW * 2);
    __bf16* Qbf   = (__bf16*)take(nX * 2);               // [B,H,T,64]
    __bf16* Kbf   = (__bf16*)take(nX * 2);               // [B,H,T,64]
    float*  Vf32  = (float*)take(nX * 4);                // [B,T,H,64]
    float*  csum  = (float*)take((size_t)BATCH * HEADS * SEQ * 4);
    __bf16* mixed = (__bf16*)take(nX * 2);
    (void)ws_size;

    const int threads = 256;

    cvt_f32_to_bf16<<<(int)((nX + threads - 1) / threads), threads, 0, stream>>>(x, Xbf, (int)nX);
    cvt_transpose_bf16<<<(int)((nW + threads - 1) / threads), threads, 0, stream>>>(Wq, WqT, DMODEL, DMODEL);
    cvt_transpose_bf16<<<(int)((nW + threads - 1) / threads), threads, 0, stream>>>(Wk, WkT, DMODEL, DMODEL);
    cvt_transpose_bf16<<<(int)((nW + threads - 1) / threads), threads, 0, stream>>>(Wv, WvT, DMODEL, DMODEL);
    cvt_transpose_bf16<<<(int)((nW + threads - 1) / threads), threads, 0, stream>>>(Wo, WoT, DMODEL, DMODEL);

    dim3 ggrid(DMODEL / 64, MROWS / 256);
    gemm_bf16_wmma<<<ggrid, 256, 0, stream>>>(Xbf, WqT, bq, nullptr, Qbf, DMODEL, 1);
    gemm_bf16_wmma<<<ggrid, 256, 0, stream>>>(Xbf, WkT, bk, nullptr, Kbf, DMODEL, 1);
    gemm_bf16_wmma<<<ggrid, 256, 0, stream>>>(Xbf, WvT, bv, Vf32, nullptr, DMODEL, 0);

    const int ncs = BATCH * HEADS * SEQ;
    zero_f32<<<(ncs + threads - 1) / threads, threads, 0, stream>>>(csum, ncs);
    dim3 agrid(SEQ / 16, HEADS, BATCH);
    attn_colsum<<<agrid, 32, 0, stream>>>(Qbf, Kbf, csum);

    mix_kernel<<<(int)((nX + threads - 1) / threads), threads, 0, stream>>>(Vf32, csum, mixed, (int)nX);
    gemm_bf16_wmma<<<ggrid, 256, 0, stream>>>(mixed, WoT, bo, (float*)d_out, nullptr, DMODEL, 0);
    (void)in_sizes; (void)n_in; (void)out_size;
}
